// NeRF_29111288332829
// MI455X (gfx1250) — matrix operand order repaired and checked by reading, not compiled, not run on hardware
//
#include <hip/hip_runtime.h>
#include <math.h>

#define N_PTS 262144
#define ROWS  128         // points per block
#define SW    320         // LDS activation stride (max layer width, padded)

static constexpr float STEP_LENGTH = 1.7320508075688772f / 1024.0f;

typedef __attribute__((ext_vector_type(16))) _Float16 v16h;
typedef __attribute__((ext_vector_type(8)))  _Float16 v8h;
typedef __attribute__((ext_vector_type(8)))  float    v8f;
typedef __attribute__((ext_vector_type(4)))  int      v4i;

// ---- packed f16 weight layout in workspace (offsets in halfs) ----
// Per (kc,nt) 32x16 B-tile: 512 halfs; lane L owns column n = nt*16+(L&15),
// k = kc*32 + (L>=16?8:0) + (t&7) + (t>=8?16:0), t=0..15 (table order). Slab for one kc = NO*32 halfs.
#define OFF_W1IN 0u           // 64x256   = 16384
#define OFF_W1H  16384u       // 3x256x256
#define OFF_W2IN 212992u      // 320x256  = 81920
#define OFF_W2H  294912u      // 3x256x256
#define OFF_WC1  491520u      // 288x128  = 36864

__global__ __launch_bounds__(256) void pack_w(const float* __restrict__ W,
                                              _Float16* __restrict__ dst,
                                              int Korig, int KP, int NO) {
  int e = blockIdx.x * blockDim.x + threadIdx.x;
  int total = KP * NO;
  if (e >= total) return;
  int blk    = e >> 9;
  int lane   = (e >> 4) & 31;
  int t      = e & 15;
  int ntiles = NO >> 4;
  int kc = blk / ntiles;
  int nt = blk - kc * ntiles;
  int n  = nt * 16 + (lane & 15);
  int k  = kc * 32 + ((lane >> 4) << 3) + (t & 7) + ((t >> 3) << 4);
  float v = (k < Korig) ? W[n * Korig + k] : 0.0f;
  dst[e] = (_Float16)v;
}

// ---- cooperative weight-slab staging: global (packed) -> LDS -----------------
#if __has_builtin(__builtin_amdgcn_global_load_async_to_lds_b128)
#define HAVE_ASYNC_LDS 1
#else
#define HAVE_ASYNC_LDS 0
#endif

template<int SLABH>   // slab size in halfs (multiple of 2048)
__device__ __forceinline__ void stage_slab(const _Float16* __restrict__ gsrc,
                                           _Float16* lds) {
  constexpr int CH = SLABH / 8;   // 16-byte chunks
#pragma unroll
  for (int c = 0; c < CH; c += 256) {
    const int idx = c + (int)threadIdx.x;           // CH % 256 == 0
    const _Float16* gp = gsrc + idx * 8;
    _Float16* lp = lds + idx * 8;
#if HAVE_ASYNC_LDS
    __builtin_amdgcn_global_load_async_to_lds_b128(
        (__attribute__((address_space(1))) v4i*)gp,
        (__attribute__((address_space(3))) v4i*)lp, 0, 0);
#else
    *(v8h*)lp = *(const v8h*)gp;
#endif
  }
}

__device__ __forceinline__ void wait_stage() {
#if HAVE_ASYNC_LDS
#if __has_builtin(__builtin_amdgcn_s_wait_asynccnt)
  __builtin_amdgcn_s_wait_asynccnt(0);
#else
  asm volatile("s_wait_asynccnt 0" ::: "memory");
#endif
#endif
  __syncthreads();
}

// ---- one fused layer: act(128xKP) @ W(KPxNO) + b -> act, in-place in LDS -----
// 8 waves: wave w -> rows {2*(w>>1), 2*(w>>1)+1} (two 16-row tiles),
//          cols (w&1)*NCT .. +NCT-1 tiles, NCT = NO/32.  B tiles come from an
//          LDS-staged, double-buffered K-slab (shared by 4 waves each).
template<int KP, int NO, bool RELU>
__device__ __forceinline__ void mlp_layer(const _Float16* __restrict__ Wf,
                                          const float* __restrict__ bias,
                                          _Float16* sAct,
                                          _Float16 (*sW)[8192]) {
  constexpr int KC   = KP / 32;
  constexpr int NCT  = NO / 32;
  constexpr int SLAB = NO * 32;
  const int tid  = threadIdx.x;
  const int wave = tid >> 5;
  const int lane = tid & 31;
  const int g    = wave & 1;
  const int p    = wave >> 1;
  const int rt0  = 2 * p, rt1 = 2 * p + 1;
  const int m    = lane & 15;
  const int hi   = lane >> 4;

  v8f acc0[NCT] = {};
  v8f acc1[NCT] = {};

  stage_slab<SLAB>(Wf, sW[0]);
  wait_stage();

  const _Float16* aRow0 = sAct + (rt0 * 16 + m) * SW;
  const _Float16* aRow1 = sAct + (rt1 * 16 + m) * SW;

  for (int kc = 0; kc < KC; ++kc) {
    if (kc + 1 < KC)
      stage_slab<SLAB>(Wf + (size_t)(kc + 1) * SLAB, sW[(kc + 1) & 1]);

    // 16-bit A 16x32 fragment: lanes 0-15 K {0..7,16..23}, lanes 16-31 K {8..15,24..31}
    const int kb = kc * 32 + hi * 8;
    v8h a0l = *(const v8h*)(aRow0 + kb);
    v8h a0h = *(const v8h*)(aRow0 + kb + 16);
    v8h a1l = *(const v8h*)(aRow1 + kb);
    v8h a1h = *(const v8h*)(aRow1 + kb + 16);
    v16h a0 = __builtin_shufflevector(a0l, a0h, 0,1,2,3,4,5,6,7,8,9,10,11,12,13,14,15);
    v16h a1 = __builtin_shufflevector(a1l, a1h, 0,1,2,3,4,5,6,7,8,9,10,11,12,13,14,15);

    const _Float16* wb = sW[kc & 1] + lane * 16;
#pragma unroll
    for (int j = 0; j < NCT; ++j) {
      v16h b = *(const v16h*)(wb + (size_t)(g * NCT + j) * 512);
      acc0[j] = __builtin_amdgcn_wmma_f32_16x16x32_f16(false, a0, false, b,
                                                       (short)0, acc0[j], false, false);
      acc1[j] = __builtin_amdgcn_wmma_f32_16x16x32_f16(false, a1, false, b,
                                                       (short)0, acc1[j], false, false);
    }
    wait_stage();   // next slab ready; all waves done reading current slab
  }

  // all activation reads complete (barrier above); write back in-place
#pragma unroll
  for (int j = 0; j < NCT; ++j) {
    const int n = (g * NCT + j) * 16 + m;
    const float bv = bias[n];
#pragma unroll
    for (int v = 0; v < 8; ++v) {
      float x0 = acc0[j][v] + bv;
      float x1 = acc1[j][v] + bv;
      if (RELU) { x0 = fmaxf(x0, 0.0f); x1 = fmaxf(x1, 0.0f); }
      sAct[(rt0 * 16 + v + hi * 8) * SW + n] = (_Float16)x0;
      sAct[(rt1 * 16 + v + hi * 8) * SW + n] = (_Float16)x1;
    }
  }
  __syncthreads();
}

__global__ __launch_bounds__(256, 1) void nerf_fused(
    const float* __restrict__ pos, const float* __restrict__ dir,
    const _Float16* __restrict__ ws,
    const float* __restrict__ b1_in, const float* __restrict__ b1_h,
    const float* __restrict__ b2_in, const float* __restrict__ b2_h,
    const float* __restrict__ w_sigma, const float* __restrict__ b_sigma,
    const float* __restrict__ b_c1, const float* __restrict__ w_c2,
    const float* __restrict__ b_c2, float* __restrict__ out) {
  __shared__ _Float16 sAct[ROWS * SW];      // 80 KB
  __shared__ _Float16 sW[2][8192];          // 32 KB double-buffered weight slab
  __shared__ float sSigma[ROWS];

  const int tid = threadIdx.x;
  const int rowBase = blockIdx.x * ROWS;

  // ---- positional encoding at cols 0..63, duplicated at 256..318 for base_mlp_2
  if (tid < ROWS) {
    const int r = rowBase + tid;
    _Float16* pr = sAct + tid * SW;
    float xyz[3] = {pos[r * 3 + 0], pos[r * 3 + 1], pos[r * 3 + 2]};
    pr[0] = (_Float16)xyz[0]; pr[1] = (_Float16)xyz[1]; pr[2] = (_Float16)xyz[2];
    for (int c = 0; c < 3; ++c) {
      float f = 1.0f;
      for (int q = 0; q < 10; ++q) {
        float v = xyz[c] * f;
        pr[3  + c * 10 + q] = (_Float16)sinf(v);
        pr[33 + c * 10 + q] = (_Float16)cosf(v);
        f *= 2.0f;
      }
    }
    pr[63] = (_Float16)0.0f;
    for (int j = 0; j < 63; ++j) pr[256 + j] = pr[j];
    pr[319] = (_Float16)0.0f;
  }
  __syncthreads();

  // ---- base_mlp_1
  mlp_layer<64, 256, true>(ws + OFF_W1IN, b1_in, sAct, sW);
  for (int i = 0; i < 3; ++i)
    mlp_layer<256, 256, true>(ws + OFF_W1H + (size_t)i * 65536u, b1_h + i * 256, sAct, sW);

  // ---- base_mlp_2 (reads 319-wide concat [h | pe])
  mlp_layer<320, 256, true>(ws + OFF_W2IN, b2_in, sAct, sW);
  for (int i = 0; i < 3; ++i)
    mlp_layer<256, 256, true>(ws + OFF_W2H + (size_t)i * 65536u, b2_h + i * 256, sAct, sW);

  // ---- sigma head + dir encoding; splice color features into cols 256..287
  if (tid < ROWS) {
    const int r = rowBase + tid;
    _Float16* pr = sAct + tid * SW;
    float acc = b_sigma[0];
    for (int k = 0; k < 256; ++k) acc += (float)pr[k] * w_sigma[k];
    sSigma[tid] = acc;

    float u[3] = {(dir[r * 3 + 0] + 1.0f) * 0.5f,
                  (dir[r * 3 + 1] + 1.0f) * 0.5f,
                  (dir[r * 3 + 2] + 1.0f) * 0.5f};
    for (int j = 0; j < 13; ++j) {       // ch0 f0..4, ch1 f0..4, ch2 f0..2
      int c = j / 5, q = j - c * 5;
      float z = u[c] * (float)(1 << q);
      pr[256 + j] = (_Float16)sinf(z);
      pr[269 + j] = (_Float16)cosf(z);
    }
    pr[282] = (_Float16)acc;
    for (int j = 283; j < 288; ++j) pr[j] = (_Float16)0.0f;
  }
  __syncthreads();

  // ---- color head layer 1: 283->288 padded, 128 wide
  mlp_layer<288, 128, true>(ws + OFF_WC1, b_c1, sAct, sW);

  // ---- color head layer 2 (128->3) + sigmoid, plus alpha
  if (tid < ROWS) {
    const int r = rowBase + tid;
    const _Float16* pr = sAct + tid * SW;
    for (int c = 0; c < 3; ++c) {
      float a = b_c2[c];
      for (int k = 0; k < 128; ++k) a += (float)pr[k] * w_c2[c * 128 + k];
      out[r * 3 + c] = 1.0f / (1.0f + expf(-a));
    }
    float s = fmaxf(sSigma[tid], 0.0f) * STEP_LENGTH;
    out[(size_t)N_PTS * 3 + r] = 1.0f - expf(-s);
  }
}

extern "C" void kernel_launch(void* const* d_in, const int* in_sizes, int n_in,
                              void* d_out, int out_size, void* d_ws, size_t ws_size,
                              hipStream_t stream) {
  const float* pos     = (const float*)d_in[0];
  const float* dirp    = (const float*)d_in[1];
  const float* w1_in   = (const float*)d_in[2];
  const float* b1_in   = (const float*)d_in[3];
  const float* w1_h    = (const float*)d_in[4];
  const float* b1_h    = (const float*)d_in[5];
  const float* w2_in   = (const float*)d_in[6];
  const float* b2_in   = (const float*)d_in[7];
  const float* w2_h    = (const float*)d_in[8];
  const float* b2_h    = (const float*)d_in[9];
  const float* w_sigma = (const float*)d_in[10];
  const float* b_sigma = (const float*)d_in[11];
  const float* w_c1    = (const float*)d_in[12];
  const float* b_c1    = (const float*)d_in[13];
  const float* w_c2    = (const float*)d_in[14];
  const float* b_c2    = (const float*)d_in[15];
  float* out = (float*)d_out;
  _Float16* ws = (_Float16*)d_ws;

  auto packs = [&](const float* W, unsigned dstOfs, int Korig, int KP, int NO) {
    int tot = KP * NO;
    pack_w<<<(tot + 255) / 256, 256, 0, stream>>>(W, ws + dstOfs, Korig, KP, NO);
  };
  packs(w1_in, OFF_W1IN, 63, 64, 256);
  for (int i = 0; i < 3; ++i) packs(w1_h + (size_t)i * 65536u, OFF_W1H + i * 65536u, 256, 256, 256);
  packs(w2_in, OFF_W2IN, 319, 320, 256);
  for (int i = 0; i < 3; ++i) packs(w2_h + (size_t)i * 65536u, OFF_W2H + i * 65536u, 256, 256, 256);
  packs(w_c1, OFF_WC1, 283, 288, 128);

  nerf_fused<<<N_PTS / ROWS, 256, 0, stream>>>(
      pos, dirp, ws, b1_in, b1_h, b2_in, b2_h,
      w_sigma, b_sigma, b_c1, w_c2, b_c2, out);
}
